// MultiHeadAttention_6236292514431
// MI455X (gfx1250) — matrix operand order repaired and checked by reading, not compiled, not run on hardware
//
#include <hip/hip_runtime.h>
#include <hip/hip_bf16.h>

// ---------------------------------------------------------------------------
// MHA: B=2, S=2048, D=1024, H=16, hd=64. fp32 in/out, bf16 WMMA internally.
// ---------------------------------------------------------------------------

typedef __attribute__((ext_vector_type(16))) __bf16 v16bf;
typedef __attribute__((ext_vector_type(8)))  __bf16 v8bf;
typedef __attribute__((ext_vector_type(8)))  float  v8f;
typedef __attribute__((ext_vector_type(4)))  unsigned int v4u;
typedef __attribute__((ext_vector_type(8)))  int v8i;
typedef __attribute__((ext_vector_type(4)))  int v4i;

#define D_EMB 1024
#define NH    16
#define HD    64
#define BATCH 2
#define SEQ   2048
#define M_TOT (BATCH * SEQ)   // 4096
#define N_QKV (3 * D_EMB)     // 3072

#define WMMA_BF16(A, B, C) \
  __builtin_amdgcn_wmma_f32_16x16x32_bf16(false, (A), false, (B), (short)0, (C), false, false)

static __device__ __forceinline__ __bf16 f2bf(float f) {
  unsigned u = __builtin_bit_cast(unsigned, f);
  unsigned r = u + 0x7fffu + ((u >> 16) & 1u);   // round-to-nearest-even
  unsigned short h = (unsigned short)(r >> 16);
  return __builtin_bit_cast(__bf16, h);
}

// A-fragment (16x32 bf16, MxK): lane<16 row=lane, k=16h+j contiguous
// caller passes p = rowBase + 16*half ; hi chunk at +8 halves.
static __device__ __forceinline__ v16bf ld_a16(const __bf16* p) {
  v8bf lo = *(const v8bf*)(p);
  v8bf hi = *(const v8bf*)(p + 8);
  return __builtin_shufflevector(lo, hi, 0,1,2,3,4,5,6,7,8,9,10,11,12,13,14,15);
}

// B-fragment (32x16 bf16, KxN): lane holds column n=lane&15, k=16*(lane>>4)+j,
// contiguous. caller passes p = colSourceRowBase + 16*(lane>>4).
static __device__ __forceinline__ v16bf ld_b16(const __bf16* p) {
  v8bf lo = *(const v8bf*)(p);
  v8bf hi = *(const v8bf*)(p + 8);
  return __builtin_shufflevector(lo, hi, 0,1,2,3,4,5,6,7,8,9,10,11,12,13,14,15);
}

static __device__ __forceinline__ float rmax16(float x) {
  x = fmaxf(x, __shfl_xor(x, 1));
  x = fmaxf(x, __shfl_xor(x, 2));
  x = fmaxf(x, __shfl_xor(x, 4));
  x = fmaxf(x, __shfl_xor(x, 8));
  return x;
}
static __device__ __forceinline__ float rsum16(float x) {
  x += __shfl_xor(x, 1);
  x += __shfl_xor(x, 2);
  x += __shfl_xor(x, 4);
  x += __shfl_xor(x, 8);
  return x;
}

// ---------------------------------------------------------------------------
__global__ __launch_bounds__(256) void cvt_f32_bf16(const float* __restrict__ in,
                                                    __bf16* __restrict__ out, int n) {
  int i = blockIdx.x * blockDim.x + threadIdx.x;
  int stride = gridDim.x * blockDim.x;
  for (; i < n; i += stride) out[i] = f2bf(in[i]);
}

// ---------------------------------------------------------------------------
// GEMM core: 32M x 64N wave tile, ping-pong double buffering (no copy movs),
// 16 WMMAs per 64-wide k-step.
// ---------------------------------------------------------------------------
static __device__ __forceinline__ void gemm_core(const __bf16* ap0, const __bf16* ap1,
                                                 const __bf16* const (&bp)[4],
                                                 v8f (&acc)[2][4]) {
  v16bf a0A = ld_a16(ap0), a1A = ld_a16(ap1);
  v16bf bA[4];
#pragma unroll
  for (int f = 0; f < 4; ++f) bA[f] = ld_b16(bp[f]);

  for (int kk = 0; kk < D_EMB; kk += 64) {
    const int k1 = kk + 32;                       // always < D_EMB
    v16bf a0B = ld_a16(ap0 + k1), a1B = ld_a16(ap1 + k1);
    v16bf bB[4];
#pragma unroll
    for (int f = 0; f < 4; ++f) bB[f] = ld_b16(bp[f] + k1);

#pragma unroll
    for (int f = 0; f < 4; ++f) {
      acc[0][f] = WMMA_BF16(a0A, bA[f], acc[0][f]);
      acc[1][f] = WMMA_BF16(a1A, bA[f], acc[1][f]);
    }

    const int k2 = (kk + 64 < D_EMB) ? kk + 64 : k1;   // clamped prefetch
    a0A = ld_a16(ap0 + k2); a1A = ld_a16(ap1 + k2);
#pragma unroll
    for (int f = 0; f < 4; ++f) bA[f] = ld_b16(bp[f] + k2);

#pragma unroll
    for (int f = 0; f < 4; ++f) {
      acc[0][f] = WMMA_BF16(a0B, bB[f], acc[0][f]);
      acc[1][f] = WMMA_BF16(a1B, bB[f], acc[1][f]);
    }
  }
}

// ---------------------------------------------------------------------------
// QKV GEMM: Y[m,n] = X[m,:] . W[n,:] + bias[n]   (M=4096, N=3072, K=1024)
// grid (M/32, N/256), 128 thr. Epilogue scatters Q (x1/8), K, V^T (bf16).
// ---------------------------------------------------------------------------
__global__ __launch_bounds__(128) void qkv_gemm(const __bf16* __restrict__ X,
                                                const __bf16* __restrict__ W,
                                                const float* __restrict__ bias,
                                                __bf16* __restrict__ Q,
                                                __bf16* __restrict__ Km,
                                                __bf16* __restrict__ VT) {
  const int lane = threadIdx.x & 31;
  const int wave = threadIdx.x >> 5;
  const int r    = lane & 15;
  const int hf   = lane >> 4;
  const int m0   = blockIdx.x * 32;
  const int nb   = blockIdx.y * 256 + wave * 64;

  const __bf16* ap0 = X + (size_t)(m0 + r) * D_EMB + 16 * hf;
  const __bf16* ap1 = ap0 + (size_t)16 * D_EMB;
  const __bf16* bp[4];
#pragma unroll
  for (int f = 0; f < 4; ++f) bp[f] = W + (size_t)(nb + f * 16 + r) * D_EMB + 16 * hf;

  v8f acc[2][4] = {{{}, {}, {}, {}}, {{}, {}, {}, {}}};
  gemm_core(ap0, ap1, bp, acc);

#pragma unroll
  for (int f = 0; f < 4; ++f) {
    const int n   = nb + f * 16 + r;
    const float bv = bias[n];
    const int sec = n >> 10;             // 0=Q 1=K 2=V
    const int nin = n & 1023;
    const int h   = nin >> 6;
    const int d   = nin & 63;
#pragma unroll
    for (int ai = 0; ai < 2; ++ai) {
#pragma unroll
      for (int e = 0; e < 8; ++e) {
        const int m = m0 + 16 * ai + e + 8 * hf;
        const int b = m >> 11;
        const int s = m & 2047;
        const float y = acc[ai][f][e] + bv;
        if (sec == 0)
          Q[((size_t)(b * NH + h) * SEQ + s) * HD + d] = f2bf(y * 0.125f);
        else if (sec == 1)
          Km[((size_t)(b * NH + h) * SEQ + s) * HD + d] = f2bf(y);
        else
          VT[((size_t)(b * NH + h) * HD + d) * SEQ + s] = f2bf(y);
      }
    }
  }
}

// ---------------------------------------------------------------------------
// Flash attention: 1 wave = one 16-row query tile of one (b,h).
// attn_step processes one 32-key tile with K frags "cur" and prefetches "nxt";
// caller alternates buffers -> no rotation movs.
// ---------------------------------------------------------------------------
static __device__ __forceinline__ void attn_step(int kt, int ktn,
                                                 const __bf16* kb, const __bf16* vb,
                                                 __bf16* pbuf, int r, int g,
                                                 const v16bf aq0, const v16bf aq1,
                                                 v16bf (&kcur)[2][2], v16bf (&knxt)[2][2],
                                                 v8f (&acc)[4],
                                                 float (&mrow)[8], float (&lrow)[8]) {
  // V fragments for this tile (issued early; consumed after softmax)
  v16bf vf[4];
#pragma unroll
  for (int f = 0; f < 4; ++f)
    vf[f] = ld_b16(vb + (size_t)(f * 16 + r) * SEQ + kt + 16 * g);

  // scores: two 16-column halves, K = 64
  v8f sc[2];
#pragma unroll
  for (int gg = 0; gg < 2; ++gg) {
    v8f c = {};
    c = WMMA_BF16(aq0, kcur[gg][0], c);
    c = WMMA_BF16(aq1, kcur[gg][1], c);
    sc[gg] = c;
  }

  // prefetch next K tile into the other buffer while softmax VALU runs
#pragma unroll
  for (int gg = 0; gg < 2; ++gg) {
    const __bf16* kr = kb + (size_t)(ktn + gg * 16 + r) * HD;
    knxt[gg][0] = ld_b16(kr + 0  + 16 * g);
    knxt[gg][1] = ld_b16(kr + 32 + 16 * g);
  }

  // online softmax update; write P (bf16) to LDS [row][key] layout
#pragma unroll
  for (int e = 0; e < 8; ++e) {
    float mx = rmax16(fmaxf(sc[0][e], sc[1][e]));
    const float mnew = fmaxf(mrow[e], mx);
    const float scale = __expf(mrow[e] - mnew);
    const float p0 = __expf(sc[0][e] - mnew);
    const float p1 = __expf(sc[1][e] - mnew);
    lrow[e] = lrow[e] * scale + rsum16(p0 + p1);
    mrow[e] = mnew;
#pragma unroll
    for (int f = 0; f < 4; ++f) acc[f][e] *= scale;
    const int row = e + 8 * g;
    pbuf[row * 32 + r]      = f2bf(p0);
    pbuf[row * 32 + 16 + r] = f2bf(p1);
  }

  asm volatile("s_wait_dscnt 0" ::: "memory");   // LDS stores visible in-wave

  // P (A-layout from LDS) @ V, K = 32
  const v16bf pa = ld_a16(pbuf + r * 32 + 16 * g);
#pragma unroll
  for (int f = 0; f < 4; ++f) acc[f] = WMMA_BF16(pa, vf[f], acc[f]);
}

__global__ __launch_bounds__(128) void attn_flash(const __bf16* __restrict__ Q,
                                                  const __bf16* __restrict__ Kmat,
                                                  const __bf16* __restrict__ VT,
                                                  __bf16* __restrict__ O) {
  __shared__ __bf16 lds_p[4][16 * 32];   // per-wave P tile, [row][key 0..31]

  const int lane = threadIdx.x & 31;
  const int wave = threadIdx.x >> 5;
  const int tile = blockIdx.x * 4 + wave;     // 0..4095
  const int qt = tile & 127;
  const int h  = (tile >> 7) & 15;
  const int b  = tile >> 11;
  const int s0 = qt * 16;
  const int r  = lane & 15;
  const int g  = lane >> 4;

  const __bf16* qb = Q    + (size_t)(b * NH + h) * SEQ * HD;
  const __bf16* kb = Kmat + (size_t)(b * NH + h) * SEQ * HD;
  const __bf16* vb = VT   + (size_t)(b * NH + h) * HD * SEQ;
  __bf16* pbuf = lds_p[wave];

  const v16bf aq0 = ld_a16(qb + (size_t)(s0 + r) * HD + 0  + 16 * g);
  const v16bf aq1 = ld_a16(qb + (size_t)(s0 + r) * HD + 32 + 16 * g);

  v8f acc[4] = {{}, {}, {}, {}};
  float mrow[8], lrow[8];
#pragma unroll
  for (int e = 0; e < 8; ++e) { mrow[e] = -3.0e38f; lrow[e] = 0.0f; }

  // K fragments for first tile into buffer A
  v16bf kfA[2][2], kfB[2][2];
#pragma unroll
  for (int gg = 0; gg < 2; ++gg) {
    const __bf16* kr = kb + (size_t)(gg * 16 + r) * HD;
    kfA[gg][0] = ld_b16(kr + 0  + 16 * g);
    kfA[gg][1] = ld_b16(kr + 32 + 16 * g);
  }

  for (int kt = 0; kt < SEQ; kt += 64) {
    attn_step(kt, kt + 32, kb, vb, pbuf, r, g, aq0, aq1, kfA, kfB, acc, mrow, lrow);
    const int ktn2 = (kt + 64 < SEQ) ? kt + 64 : kt + 32;   // clamped prefetch
    attn_step(kt + 32, ktn2, kb, vb, pbuf, r, g, aq0, aq1, kfB, kfA, acc, mrow, lrow);
  }

  // normalize and store attn output bf16 in [B, S, H*hd] layout
#pragma unroll
  for (int f = 0; f < 4; ++f) {
#pragma unroll
    for (int e = 0; e < 8; ++e) {
      const int row = e + 8 * g;
      const int col = f * 16 + r;
      const float y = acc[f][e] / lrow[e];
      O[((size_t)(b * SEQ + s0 + row)) * D_EMB + h * HD + col] = f2bf(y);
    }
  }
}

// ---------------------------------------------------------------------------
// Output GEMM: out[m,n] = A[m,:] . W[n,:] + bias[n]  (M=4096, N=1024, K=1024)
// ---------------------------------------------------------------------------
__global__ __launch_bounds__(128) void out_gemm(const __bf16* __restrict__ A,
                                                const __bf16* __restrict__ W,
                                                const float* __restrict__ bias,
                                                float* __restrict__ out) {
  const int lane = threadIdx.x & 31;
  const int wave = threadIdx.x >> 5;
  const int r    = lane & 15;
  const int hf   = lane >> 4;
  const int m0   = blockIdx.x * 32;
  const int nb   = blockIdx.y * 256 + wave * 64;

  const __bf16* ap0 = A + (size_t)(m0 + r) * D_EMB + 16 * hf;
  const __bf16* ap1 = ap0 + (size_t)16 * D_EMB;
  const __bf16* bp[4];
#pragma unroll
  for (int f = 0; f < 4; ++f) bp[f] = W + (size_t)(nb + f * 16 + r) * D_EMB + 16 * hf;

  v8f acc[2][4] = {{{}, {}, {}, {}}, {{}, {}, {}, {}}};
  gemm_core(ap0, ap1, bp, acc);

#pragma unroll
  for (int f = 0; f < 4; ++f) {
    const int n = nb + f * 16 + r;
    const float bv = bias[n];
#pragma unroll
    for (int ai = 0; ai < 2; ++ai) {
#pragma unroll
      for (int e = 0; e < 8; ++e) {
        const int m = m0 + 16 * ai + e + 8 * hf;
        out[(size_t)m * D_EMB + n] = acc[ai][f][e] + bv;
      }
    }
  }
}

// ---------------------------------------------------------------------------
// TDM demo / L2 warm: stream 64x64 bf16 tiles through LDS with the Tensor
// Data Mover (tensor_load_to_lds + s_wait_tensorcnt). Data is never read;
// this only pulls K/V lines through L2 ahead of attention. 1 wave per block.
// D# layout per CDNA5 ISA (2D, groups 2/3 zero).
// clang-23 lane: 6-arg builtin (extra int32x8 group).
// ---------------------------------------------------------------------------
__global__ __launch_bounds__(32) void tdm_l2_warm(const __bf16* __restrict__ src) {
  __shared__ __bf16 scratch[64 * 64];   // 8 KB landing pad (dead store)

  const unsigned long long ga =
      (unsigned long long)(const void*)(src + (size_t)blockIdx.x * 64 * 64);
  const unsigned lds = (unsigned)(unsigned long long)(const void*)&scratch[0];

  v4u g0;
  g0[0] = 1u;                                    // count=1, user descriptor
  g0[1] = lds;                                   // lds_addr
  g0[2] = (unsigned)ga;                          // global_addr[31:0]
  g0[3] = (unsigned)((ga >> 32) & 0x1FFFFFFu)    // global_addr[56:32]
          | (2u << 30);                          // type=2 ("image")

  v8i g1;
  g1[0] = 0x00010000;          // workgroup_mask=0, data_size=1 (2B)
  g1[1] = 64 << 16;            // tensor_dim0[15:0]=64 (barrier addr=0)
  g1[2] = 64 << 16;            // tensor_dim0 hi=0, tensor_dim1[15:0]=64
  g1[3] = 64 << 16;            // tensor_dim1 hi=0, tile_dim0=64
  g1[4] = 64;                  // tile_dim1=64, tile_dim2=0
  g1[5] = 64;                  // tensor_dim0_stride=64
  g1[6] = 0;
  g1[7] = 0;

  v4i gz4 = {0, 0, 0, 0};
  v8i gz8 = {0, 0, 0, 0, 0, 0, 0, 0};
  __builtin_amdgcn_tensor_load_to_lds(g0, g1, gz4, gz4, gz8, 0);
  __builtin_amdgcn_s_wait_tensorcnt(0);
}

// ---------------------------------------------------------------------------
extern "C" void kernel_launch(void* const* d_in, const int* in_sizes, int n_in,
                              void* d_out, int out_size, void* d_ws, size_t ws_size,
                              hipStream_t stream) {
  (void)in_sizes; (void)n_in; (void)out_size; (void)ws_size;

  const float* x     = (const float*)d_in[0];
  const float* qkv_w = (const float*)d_in[1];
  const float* qkv_b = (const float*)d_in[2];
  const float* out_w = (const float*)d_in[3];
  const float* out_b = (const float*)d_in[4];

  char* ws = (char*)d_ws;
  size_t o = 0;
  auto take = [&](size_t bytes) { char* p = ws + o; o += (bytes + 255) & ~size_t(255); return p; };

  __bf16* xb  = (__bf16*)take((size_t)M_TOT * D_EMB * 2);        // 8 MB
  __bf16* wb  = (__bf16*)take((size_t)N_QKV * D_EMB * 2);        // 6 MB
  __bf16* owb = (__bf16*)take((size_t)D_EMB * D_EMB * 2);        // 2 MB
  __bf16* Qb  = (__bf16*)take((size_t)BATCH * NH * SEQ * HD * 2);// 8 MB
  __bf16* Kb  = (__bf16*)take((size_t)BATCH * NH * SEQ * HD * 2);// 8 MB
  __bf16* VTb = (__bf16*)take((size_t)BATCH * NH * HD * SEQ * 2);// 8 MB
  __bf16* ATT = (__bf16*)take((size_t)M_TOT * D_EMB * 2);        // 8 MB

  const int n_x = M_TOT * D_EMB, n_w = N_QKV * D_EMB, n_ow = D_EMB * D_EMB;
  cvt_f32_bf16<<<(n_x  + 255) / 256, 256, 0, stream>>>(x,     xb,  n_x);
  cvt_f32_bf16<<<(n_w  + 255) / 256, 256, 0, stream>>>(qkv_w, wb,  n_w);
  cvt_f32_bf16<<<(n_ow + 255) / 256, 256, 0, stream>>>(out_w, owb, n_ow);

  qkv_gemm<<<dim3(M_TOT / 32, N_QKV / 256), 128, 0, stream>>>(xb, wb, qkv_b, Qb, Kb, VTb);

  // TDM warm of K + V^T (16 MB contiguous starting at Kb): 2048 x 8KB tiles
  tdm_l2_warm<<<2048, 32, 0, stream>>>(Kb);

  const int tiles = BATCH * NH * (SEQ / 16);            // 4096 waves
  attn_flash<<<tiles / 4, 128, 0, stream>>>(Qb, Kb, VTb, ATT);

  out_gemm<<<dim3(M_TOT / 32, D_EMB / 256), 128, 0, stream>>>(ATT, owb, out_b, (float*)d_out);
}
